// NeuralIID_8186207666603
// MI455X (gfx1250) — matrix operand order repaired and checked by reading, not compiled, not run on hardware
//
#include <hip/hip_runtime.h>
#include <hip/hip_bf16.h>

// ---------------- problem constants ----------------
#define BB 16
#define TT 256
#define DD 768
#define SS 10
#define HH 37
#define OO 37
#define MDIM (BB*TT)            // 4096
#define NEMISS (SS*HH*OO)       // 13690
#define NHID HH                 // 37
#define NTOT (NEMISS + NHID)    // 13727
#define NPAD 13824              // 108 * 128
#define KDIM DD                 // 768
#define EMISS_W 0.7f

// GEMM tiling
#define BM 64
#define BN 128
#define BK 32

typedef __attribute__((ext_vector_type(16))) __bf16 v16bf;
typedef __attribute__((ext_vector_type(8)))  float  v8f;
typedef __attribute__((ext_vector_type(4)))  unsigned int u32x4;
typedef __attribute__((ext_vector_type(8)))  int          i32x8;
typedef __attribute__((ext_vector_type(4)))  int          i32x4;

union FragAB { v16bf v; uint4 q[2]; };

#if defined(__has_builtin)
#  if __has_builtin(__builtin_amdgcn_tensor_load_to_lds)
#    define HAVE_TDM 1
#  else
#    define HAVE_TDM 0
#  endif
#else
#  define HAVE_TDM 0
#endif

__device__ __forceinline__ unsigned short f2bf_rne(float f) {
    unsigned int u = __float_as_uint(f);
    unsigned int r = u + 0x7FFFu + ((u >> 16) & 1u);
    return (unsigned short)(r >> 16);
}

#if HAVE_TDM
// Issue one 2D TDM tile load: tile_d1 rows x tile_d0 elements (2-byte data),
// row stride = stride_elems, packed contiguously into LDS at lds_addr.
// Descriptor layout per CDNA5 ISA 8.3-8.6 (groups 2/3 zero => 2D tensor).
// This toolchain (clang-23 lane) uses the 6-arg builtin form:
//   (u32x4 g0, i32x8 g1, i32x4 g2, i32x4 g3, i32x8 extra, i32 cpol)
__device__ __forceinline__ void tdm_load_2d(unsigned lds_addr, const void* gptr,
                                            int tile_d0, int tile_d1,
                                            int stride_elems) {
    unsigned long long ga = (unsigned long long)(size_t)gptr;
    u32x4 g0;
    g0[0] = 1u;                                    // count=1, user mode
    g0[1] = lds_addr;                              // LDS byte address
    g0[2] = (unsigned)(ga & 0xFFFFFFFFu);          // global_addr[31:0]
    g0[3] = (unsigned)(ga >> 32) | (2u << 30);     // global_addr[56:32] | type=2
    i32x8 g1;
    g1[0] = 1 << 16;                               // data_size=1 (2 bytes)
    g1[1] = (tile_d0 & 0xFFFF) << 16;              // tensor_dim0[15:0] (== tile_d0)
    g1[2] = (tile_d1 & 0xFFFF) << 16;              // tensor_dim0 hi=0 | tensor_dim1 lo
    g1[3] = (tile_d0 & 0xFFFF) << 16;              // tensor_dim1 hi=0 | tile_dim0
    g1[4] = (tile_d1 & 0xFFFF);                    // tile_dim1 | tile_dim2=0
    g1[5] = stride_elems;                          // tensor_dim0_stride[31:0]
    g1[6] = 0;                                     // stride0 hi | stride1 lo
    g1[7] = 0;                                     // stride1 hi
    i32x4 z4 = {0, 0, 0, 0};
    i32x8 z8 = {0, 0, 0, 0, 0, 0, 0, 0};
    __builtin_amdgcn_tensor_load_to_lds(g0, g1, z4, z4, z8, 0);
}
#endif

// ---------------- prep kernels ----------------
__global__ void k_conv_emb(const float* __restrict__ emb,
                           unsigned short* __restrict__ emb_bf) {
    int i = blockIdx.x * blockDim.x + threadIdx.x;   // < MDIM*KDIM
    emb_bf[i] = f2bf_rne(emb[i]);
}

__global__ void k_conv_wcat(const float* __restrict__ We,   // [13690][768]
                            const float* __restrict__ Wh,   // [37][768]
                            unsigned short* __restrict__ Wcat) { // [13824][768]
    int i = blockIdx.x * blockDim.x + threadIdx.x;   // < NPAD*KDIM
    int row = i / KDIM;
    int k   = i - row * KDIM;
    float v = 0.0f;
    if (row < NEMISS)      v = We[(size_t)row * KDIM + k];
    else if (row < NTOT)   v = Wh[(row - NEMISS) * KDIM + k];
    Wcat[i] = f2bf_rne(v);
}

__global__ void k_obs_argmax(const float* __restrict__ obs,  // [40960][37]
                             int* __restrict__ obs_idx) {
    int i = blockIdx.x * blockDim.x + threadIdx.x;   // < MDIM*SS
    const float* p = obs + (size_t)i * OO;
    int sel = 0;
    for (int o = 0; o < OO; ++o) if (p[o] > 0.5f) sel = o;
    obs_idx[i] = sel;
}

__global__ void k_estat(const float* __restrict__ unn,   // [37][37]
                        float* __restrict__ estat) {     // 0.3*softmax rows
    int i = blockIdx.x * blockDim.x + threadIdx.x;
    if (i >= HH * OO) return;
    int h = i / OO, o = i - h * OO;
    const float* row = unn + h * OO;
    float m = -1e30f;
    for (int j = 0; j < OO; ++j) m = fmaxf(m, row[j]);
    float s = 0.0f;
    for (int j = 0; j < OO; ++j) s += __expf(row[j] - m);
    estat[i] = (1.0f - EMISS_W) * (__expf(row[o] - m) / s);
}

// ---------------- main GEMM (bf16 WMMA + TDM staging) ----------------
__global__ __launch_bounds__(256)
void k_gemm_wmma(const unsigned short* __restrict__ A,    // emb_bf [4096][768]
                 const unsigned short* __restrict__ Bw,   // Wcat   [13824][768]
                 const float* __restrict__ be,            // [13690]
                 const float* __restrict__ bh,            // [37]
                 float* __restrict__ outE,                // d_out+1 [4096][13690]
                 float* __restrict__ outH) {              // [4096][37]
    __shared__ __align__(16) unsigned short As[2][BM][BK];
    __shared__ __align__(16) unsigned short Bs[2][BN][BK];

    const int tid   = threadIdx.x;
    const int lane  = tid & 31;
    const int wave  = tid >> 5;          // 0..7
    const int waveM = wave >> 2;         // 0..1
    const int waveN = wave & 3;          // 0..3
    const int blockM = blockIdx.y * BM;
    const int blockN = blockIdx.x * BN;

    v8f c[2][2] = {};

    const int lo = (lane < 16) ? 0 : 1;  // K-half select per ISA fragment layout
    const int ml = lane & 15;

#if HAVE_TDM
    // LDS aperture: low 32 bits of the flat shared-memory address = LDS offset
    const unsigned ldsA = (unsigned)(size_t)(void*)&As[0][0][0];
    const unsigned ldsB = (unsigned)(size_t)(void*)&Bs[0][0][0];
    const bool issuer = (wave == 0);
    if (issuer) {   // preload buffer 0 via Tensor Data Mover
        tdm_load_2d(ldsA, A  + (size_t)blockM * KDIM, BK, BM, KDIM);
        tdm_load_2d(ldsB, Bw + (size_t)blockN * KDIM, BK, BN, KDIM);
    }
#else
    const int arow = tid >> 2;           // 0..63
    const int acol = (tid & 3) * 8;      // shorts
    const int brow = tid >> 1;           // 0..127
    const int bcol = (tid & 1) * 16;     // shorts
    {
        const uint4* ag = (const uint4*)(A + (size_t)(blockM + arow) * KDIM + acol);
        *(uint4*)&As[0][arow][acol] = ag[0];
        const uint4* bg = (const uint4*)(Bw + (size_t)(blockN + brow) * KDIM + bcol);
        *(uint4*)&Bs[0][brow][bcol]     = bg[0];
        *(uint4*)&Bs[0][brow][bcol + 8] = bg[1];
    }
    __syncthreads();
#endif

    int buf = 0;
    for (int k = 0; k < KDIM; k += BK) {
        int nk = k + BK;
#if HAVE_TDM
        __syncthreads();   // all waves done computing on buf^1 before TDM refills it
        if (issuer) {
            if (nk < KDIM) {
                tdm_load_2d(ldsA + (unsigned)((buf ^ 1) * (BM * BK * 2)),
                            A + (size_t)blockM * KDIM + nk, BK, BM, KDIM);
                tdm_load_2d(ldsB + (unsigned)((buf ^ 1) * (BN * BK * 2)),
                            Bw + (size_t)blockN * KDIM + nk, BK, BN, KDIM);
                __builtin_amdgcn_s_wait_tensorcnt(2);  // current buf's pair done
            } else {
                __builtin_amdgcn_s_wait_tensorcnt(0);
            }
        }
        __syncthreads();   // publish buf to all waves
#else
        if (nk < KDIM) {
            const uint4* ag = (const uint4*)(A + (size_t)(blockM + arow) * KDIM + nk + acol);
            *(uint4*)&As[buf ^ 1][arow][acol] = ag[0];
            const uint4* bg = (const uint4*)(Bw + (size_t)(blockN + brow) * KDIM + nk + bcol);
            *(uint4*)&Bs[buf ^ 1][brow][bcol]     = bg[0];
            *(uint4*)&Bs[buf ^ 1][brow][bcol + 8] = bg[1];
        }
#endif

        FragAB fa[2], fb[2];
#pragma unroll
        for (int i = 0; i < 2; ++i) {
            int r = waveM * 32 + i * 16 + ml;        // A: lane = M row
            fa[i].q[0] = *(const uint4*)&As[buf][r][lo * 8];        // K 0-7 / 8-15
            fa[i].q[1] = *(const uint4*)&As[buf][r][16 + lo * 8];   // K 16-23 / 24-31
        }
#pragma unroll
        for (int j = 0; j < 2; ++j) {
            int n = waveN * 32 + j * 16 + ml;        // B: lane = N column
            fb[j].q[0] = *(const uint4*)&Bs[buf][n][lo * 8];        // K 0-7 / 8-15
            fb[j].q[1] = *(const uint4*)&Bs[buf][n][16 + lo * 8];   // K 16-23 / 24-31
        }
#pragma unroll
        for (int i = 0; i < 2; ++i)
#pragma unroll
            for (int j = 0; j < 2; ++j)
                c[i][j] = __builtin_amdgcn_wmma_f32_16x16x32_bf16(
                    false, fa[i].v, false, fb[j].v,
                    (short)0, c[i][j], false, false);

#if !HAVE_TDM
        __syncthreads();
#endif
        buf ^= 1;
    }

    // epilogue: C/D layout -> lanes 0-15: M = p, N = lane; lanes 16-31: M = p+8
    const int mh = (lane >> 4) * 8;
#pragma unroll
    for (int i = 0; i < 2; ++i) {
#pragma unroll
        for (int j = 0; j < 2; ++j) {
            int ng = blockN + waveN * 32 + j * 16 + ml;
#pragma unroll
            for (int p = 0; p < 8; ++p) {
                int mg = blockM + waveM * 32 + i * 16 + mh + p;
                float v = c[i][j][p];
                if (ng < NEMISS) {
                    outE[(size_t)mg * NEMISS + ng] = v + be[ng];
                } else if (ng < NTOT) {
                    outH[mg * NHID + (ng - NEMISS)] = v + bh[ng - NEMISS];
                }
            }
        }
    }
}

// ---------------- softmax over O + log-mix, in place ----------------
__global__ void k_logemiss(float* __restrict__ outE,         // [4096*10*37][37]
                           const float* __restrict__ estat) { // 0.3*emiss [37][37]
    int r = blockIdx.x * blockDim.x + threadIdx.x;   // < MDIM*SS*HH
    float* p = outE + (size_t)r * OO;
    int h = r % HH;
    float x[OO];
    float m = -1e30f;
    for (int o = 0; o < OO; ++o) { x[o] = p[o]; m = fmaxf(m, x[o]); }
    float s = 0.0f;
    for (int o = 0; o < OO; ++o) { float e = __expf(x[o] - m); x[o] = e; s += e; }
    float inv = EMISS_W / s;
    const float* es = estat + h * OO;
    for (int o = 0; o < OO; ++o) p[o] = __logf(es[o] + x[o] * inv);
}

// ---------------- gather observed column, sum over S (deterministic) ---------
__global__ void k_gather(const float* __restrict__ outE,    // log_emiss
                         const int* __restrict__ obs_idx,   // [4096*10]
                         float* __restrict__ acc) {         // [4096][37]
    int i = blockIdx.x * blockDim.x + threadIdx.x;  // < MDIM*HH
    int bt = i / HH, h = i - bt * HH;
    float s = 0.0f;
    for (int sidx = 0; sidx < SS; ++sidx) {
        int col = obs_idx[bt * SS + sidx];
        s += outE[(size_t)bt * NEMISS + sidx * (HH * OO) + h * OO + col];
    }
    acc[i] = s;
}

// ---------------- final logsumexp + ragged reduction (one WG) ----------------
__global__ __launch_bounds__(256)
void k_final(const float* __restrict__ hid,     // [4096][37]
             const float* __restrict__ acc,     // [4096][37]
             const int* __restrict__ seql,      // [16]
             float* __restrict__ out0) {
    __shared__ float red[256];
    int tid = threadIdx.x;
    float local = 0.0f;
    for (int bt = tid; bt < MDIM; bt += 256) {
        const float* hx = hid + bt * HH;
        const float* ax = acc + bt * HH;
        float m1 = -1e30f;
        for (int h = 0; h < HH; ++h) m1 = fmaxf(m1, hx[h]);
        float s1 = 0.0f;
        for (int h = 0; h < HH; ++h) s1 += __expf(hx[h] - m1);
        float lse = m1 + __logf(s1);            // log-softmax normalizer
        float y[HH];
        float m2 = -1e30f;
        for (int h = 0; h < HH; ++h) { y[h] = (hx[h] - lse) + ax[h]; m2 = fmaxf(m2, y[h]); }
        float s2 = 0.0f;
        for (int h = 0; h < HH; ++h) s2 += __expf(y[h] - m2);
        float marg = m2 + __logf(s2);
        int b = bt >> 8, t = bt & 255;
        if (t < seql[b]) local += marg;
    }
    red[tid] = local;
    __syncthreads();
    for (int s = 128; s > 0; s >>= 1) {
        if (tid < s) red[tid] += red[tid + s];
        __syncthreads();
    }
    if (tid == 0) out0[0] = red[0] * (1.0f / (float)BB);
}

// ---------------- launcher ----------------
extern "C" void kernel_launch(void* const* d_in, const int* in_sizes, int n_in,
                              void* d_out, int out_size, void* d_ws, size_t ws_size,
                              hipStream_t stream) {
    const float* emb  = (const float*)d_in[0];
    const float* obs  = (const float*)d_in[1];
    const int*   seql = (const int*)  d_in[2];
    const float* Wh   = (const float*)d_in[3];
    const float* bh   = (const float*)d_in[4];
    const float* We   = (const float*)d_in[5];
    const float* be   = (const float*)d_in[6];
    const float* unn  = (const float*)d_in[7];

    float* out  = (float*)d_out;       // out[0] = log_likelihood
    float* outE = out + 1;             // [4096][13690] log_emiss

    char* ws = (char*)d_ws;
    size_t off = 0;
    unsigned short* emb_bf = (unsigned short*)(ws + off); off += (size_t)MDIM * KDIM * 2;      // 6.3 MB
    unsigned short* wcat   = (unsigned short*)(ws + off); off += (size_t)NPAD * KDIM * 2;      // 21.2 MB
    float* hid   = (float*)(ws + off); off += (size_t)MDIM * NHID * 4;                         // 606 KB
    float* acc   = (float*)(ws + off); off += (size_t)MDIM * NHID * 4;                         // 606 KB
    int*   oidx  = (int*)  (ws + off); off += (size_t)MDIM * SS * 4;                           // 164 KB
    float* estat = (float*)(ws + off); off += (size_t)HH * OO * 4;                             // 5.5 KB

    // 1) prep
    k_conv_emb  <<<(MDIM * KDIM) / 256, 256, 0, stream>>>(emb, emb_bf);
    k_conv_wcat <<<(NPAD * KDIM) / 256, 256, 0, stream>>>(We, Wh, wcat);
    k_obs_argmax<<<(MDIM * SS) / 256, 256, 0, stream>>>(obs, oidx);
    k_estat     <<<(HH * OO + 255) / 256, 256, 0, stream>>>(unn, estat);

    // 2) fused GEMM (emission + hidden logits) via bf16 WMMA, TDM tile staging
    dim3 grid(NPAD / BN, MDIM / BM);   // 108 x 64
    k_gemm_wmma<<<grid, 256, 0, stream>>>(emb_bf, wcat, be, bh, outE, hid);

    // 3) softmax over O + log-mix, in place in d_out
    k_logemiss<<<(MDIM * SS * HH) / 256, 256, 0, stream>>>(outE, estat);

    // 4) gather observed column, sum over sources
    k_gather<<<(MDIM * HH) / 256, 256, 0, stream>>>(outE, oidx, acc);

    // 5) final marginal + ragged reduction -> scalar
    k_final<<<1, 256, 0, stream>>>(hid, acc, seql, out);
}
